// Block_47614007443572
// MI455X (gfx1250) — compile-verified, matrix operand order checked by reading, not run on hardware
//
#include <hip/hip_runtime.h>
#include <hip/hip_bf16.h>
#include <math.h>

// ---------------- problem constants ----------------
constexpr int Bc = 2, Tc = 2048, Cc = 1024, Hc = 16, Dc = 64;
constexpr int Mrows = Bc * Tc;          // 4096 rows for all activation GEMMs

typedef __bf16 v16bf __attribute__((ext_vector_type(16)));
typedef float  v8f   __attribute__((ext_vector_type(8)));
typedef unsigned int u32x4 __attribute__((ext_vector_type(4)));
typedef unsigned int u32x8 __attribute__((ext_vector_type(8)));

// ---------------- small helpers ----------------
__device__ __forceinline__ unsigned short f2bf(float f) {
    unsigned u = __builtin_bit_cast(unsigned, f);
    u += 0x7FFFu + ((u >> 16) & 1u);          // round-to-nearest-even
    return (unsigned short)(u >> 16);
}

// Load one bf16 A/B fragment (16x32 / 32x16, K contiguous per lane).
// Layout per CDNA5 ISA 7.12.2:
//   elems 0..7  <- K = kbl .. kbl+7
//   elems 8..15 <- K = 16+kbl .. 23+kbl        (kbl = lane>=16 ? 8 : 0)
__device__ __forceinline__ v16bf ld_frag(const unsigned short* p, int kbl) {
    v16bf f;
    uint4* fp = reinterpret_cast<uint4*>(&f);
    fp[0] = *reinterpret_cast<const uint4*>(p + kbl);
    fp[1] = *reinterpret_cast<const uint4*>(p + kbl + 16);
    return f;
}

__device__ __forceinline__ float redmax16(float v) {
    #pragma unroll
    for (int m = 1; m < 16; m <<= 1) v = fmaxf(v, __shfl_xor(v, m, 16));
    return v;
}
__device__ __forceinline__ float redsum16(float v) {
    #pragma unroll
    for (int m = 1; m < 16; m <<= 1) v += __shfl_xor(v, m, 16);
    return v;
}

// Explicit in-wave LDS sequencing point (DS ops are in-order per wave; this
// also stops the compiler from reordering LDS accesses around it).
__device__ __forceinline__ void lds_fence() {
    asm volatile("s_wait_dscnt 0x0" ::: "memory");
}

// ---- Tensor Data Mover: load a 2D tile (rows x cols of bf16, row-major,
// row stride = tensor elems) from global into LDS with 16B padding every
// 64B row (-> LDS row stride 40 elems).  Issue from ONE wave only.
__device__ __forceinline__ void tdm_load_tile_bf16(
        const unsigned short* gsrc, unsigned lds_byte_addr,
        unsigned tile_cols /*=32*/, unsigned tile_rows /*=128*/,
        unsigned tensor_cols, unsigned tensor_rows) {
    const unsigned long long ga = (unsigned long long)(const void*)gsrc;
    u32x4 g0;
    g0[0] = 1u;                               // count=1, user descriptor
    g0[1] = lds_byte_addr;                    // lds_addr [63:32]
    g0[2] = (unsigned)ga;                     // global_addr lo
    g0[3] = (unsigned)((ga >> 32) & 0x01FFFFFFu) | (2u << 30);  // addr hi | type=2
    u32x8 g1;
    // data_size=1 (2B) | pad_enable | pad_interval=3 (16 DW = 64B) |
    // pad_amount=3 (4 DW = 16B)
    g1[0] = (1u << 16) | (1u << 20) | (3u << 22) | (3u << 25);
    g1[1] = (tensor_cols & 0xFFFFu) << 16;                    // tensor_dim0 lo
    g1[2] = (tensor_cols >> 16) | ((tensor_rows & 0xFFFFu) << 16);
    g1[3] = (tensor_rows >> 16) | (tile_cols << 16);          // tile_dim0
    g1[4] = tile_rows;                                        // tile_dim1
    g1[5] = tensor_cols;                                      // dim0_stride lo
    g1[6] = 0u;
    g1[7] = 0u;
    asm volatile("tensor_load_to_lds %0, %1" :: "s"(g0), "s"(g1) : "memory");
}

// ---------------- f32 -> bf16 weight convert ----------------
__global__ void to_bf16_kernel(const float* __restrict__ in,
                               unsigned short* __restrict__ out, size_t n) {
    size_t i = (size_t)blockIdx.x * blockDim.x + threadIdx.x;
    if (i < n) out[i] = f2bf(in[i]);
}

// ---------------- LayerNorm (row of C=1024) -> bf16 ----------------
__global__ __launch_bounds__(256)
void layernorm_bf16_kernel(const float* __restrict__ x,
                           const float* __restrict__ w,
                           const float* __restrict__ b,
                           unsigned short* __restrict__ out) {
    __shared__ float red[256];
    const int row = blockIdx.x;
    const int tid = threadIdx.x;
    const float* xr = x + (size_t)row * Cc;
    float4 v = *reinterpret_cast<const float4*>(xr + tid * 4);
    float s = v.x + v.y + v.z + v.w;
    red[tid] = s; __syncthreads();
    for (int st = 128; st > 0; st >>= 1) {
        if (tid < st) red[tid] += red[tid + st];
        __syncthreads();
    }
    const float mu = red[0] * (1.0f / Cc);
    __syncthreads();
    float dx = v.x - mu, dy = v.y - mu, dz = v.z - mu, dw = v.w - mu;
    red[tid] = dx * dx + dy * dy + dz * dz + dw * dw; __syncthreads();
    for (int st = 128; st > 0; st >>= 1) {
        if (tid < st) red[tid] += red[tid + st];
        __syncthreads();
    }
    const float rstd = rsqrtf(red[0] * (1.0f / Cc) + 1e-5f);
    const int c0 = tid * 4;
    float4 wv = *reinterpret_cast<const float4*>(w + c0);
    float4 bv = *reinterpret_cast<const float4*>(b + c0);
    unsigned short* o = out + (size_t)row * Cc + c0;
    o[0] = f2bf(dx * rstd * wv.x + bv.x);
    o[1] = f2bf(dy * rstd * wv.y + bv.y);
    o[2] = f2bf(dz * rstd * wv.z + bv.z);
    o[3] = f2bf(dw * rstd * wv.w + bv.w);
}

// ---------------- WMMA GEMM:  out[M,N] = A[M,K](bf16) * B[K,N](bf16) + bias ----
// Block: 256 threads (8 waves), 128x128 tile; wave = 64x32 = 4x2 WMMA tiles.
// A tile staged by the Tensor Data Mover (wave 0), B tile transposed by VALU.
// EPI: 0 = f32, 1 = bf16, 2 = gelu->bf16, 3 = f32 + residual
template <int EPI>
__global__ __launch_bounds__(256)
void gemm_bf16_wmma(const unsigned short* __restrict__ A,
                    const unsigned short* __restrict__ Bw,
                    const float* __restrict__ bias,
                    const float* __restrict__ res,
                    void* __restrict__ outp,
                    int M, int N, int K) {
    __shared__ __align__(16) unsigned short As[128 * 40];   // [row][k] pad 8
    __shared__ __align__(16) unsigned short Bt[128 * 40];   // [n][k]   pad 8
    const int tid  = threadIdx.x;
    const int lane = tid & 31;
    const int wave = tid >> 5;
    const int wm   = wave >> 2;           // 0..1
    const int wn   = wave & 3;            // 0..3
    const int m0   = blockIdx.y * 128;
    const int n0   = blockIdx.x * 128;
    const int lrow = lane & 15;
    const int kbl  = (lane >= 16) ? 8 : 0;
    const unsigned as_lds = (unsigned)(size_t)(const void*)&As[0];

    v8f acc[4][2] = {};

    for (int k0 = 0; k0 < K; k0 += 32) {
        // --- stage A tile 128x32 via TDM (one DMA, wave 0 only) ---
        if (wave == 0) {
            tdm_load_tile_bf16(&A[(size_t)m0 * K + k0], as_lds,
                               32u, 128u, (unsigned)K, (unsigned)M);
            __builtin_amdgcn_s_wait_tensorcnt(0);
        }
        // --- stage B tile 32x128 transposed -> Bt[n][k] ---
        {
            int kk = tid >> 3, ng = tid & 7;
            const unsigned short* src = &Bw[(size_t)(k0 + kk) * N + n0 + ng * 16];
            union { uint4 q4[2]; unsigned short e[16]; } bb;
            bb.q4[0] = *reinterpret_cast<const uint4*>(src);
            bb.q4[1] = *reinterpret_cast<const uint4*>(src + 8);
            #pragma unroll
            for (int j = 0; j < 16; ++j) Bt[(ng * 16 + j) * 40 + kk] = bb.e[j];
        }
        if (k0 + 32 < K)   // prefetch next B tile row for this thread
            __builtin_prefetch(&Bw[(size_t)(k0 + 32 + (tid >> 3)) * N + n0], 0, 0);
        __syncthreads();

        v16bf af[4], bfm[2];
        #pragma unroll
        for (int mi = 0; mi < 4; ++mi)
            af[mi] = ld_frag(&As[(wm * 64 + mi * 16 + lrow) * 40], kbl);
        #pragma unroll
        for (int ni = 0; ni < 2; ++ni)
            bfm[ni] = ld_frag(&Bt[(wn * 32 + ni * 16 + lrow) * 40], kbl);

        #pragma unroll
        for (int mi = 0; mi < 4; ++mi)
            #pragma unroll
            for (int ni = 0; ni < 2; ++ni)
                acc[mi][ni] = __builtin_amdgcn_wmma_f32_16x16x32_bf16(
                    false, af[mi], false, bfm[ni], (short)0, acc[mi][ni],
                    false, false);
        __syncthreads();
    }

    // epilogue — C layout: row = r + (lane>=16 ? 8 : 0), col = lane&15
    const int hoff = (lane >= 16) ? 8 : 0;
    #pragma unroll
    for (int mi = 0; mi < 4; ++mi) {
        #pragma unroll
        for (int ni = 0; ni < 2; ++ni) {
            const int col = n0 + wn * 32 + ni * 16 + lrow;
            const float bval = bias[col];
            #pragma unroll
            for (int r = 0; r < 8; ++r) {
                const int row = m0 + wm * 64 + mi * 16 + r + hoff;
                const size_t o = (size_t)row * N + col;
                float vv = acc[mi][ni][r] + bval;
                if (EPI == 0) {
                    ((float*)outp)[o] = vv;
                } else if (EPI == 1) {
                    ((unsigned short*)outp)[o] = f2bf(vv);
                } else if (EPI == 2) {
                    vv = 0.5f * vv * (1.0f + erff(vv * 0.70710678118654752f));
                    ((unsigned short*)outp)[o] = f2bf(vv);
                } else {
                    ((float*)outp)[o] = res[o] + vv;
                }
            }
        }
    }
}

// ---------------- QKV split: [B,T,3C] bf16 -> q/k/v [B,H,T,D] bf16 ----------
__global__ void split_qkv_kernel(const unsigned short* __restrict__ qkv,
                                 unsigned short* __restrict__ q,
                                 unsigned short* __restrict__ k,
                                 unsigned short* __restrict__ v) {
    size_t idx = (size_t)blockIdx.x * blockDim.x + threadIdx.x;  // < B*T*3C
    const int c3 = (int)(idx % (3 * Cc));
    const int t  = (int)((idx / (3 * Cc)) % Tc);
    const int b  = (int)(idx / ((size_t)(3 * Cc) * Tc));
    const int which = c3 / Cc;
    const int hd = c3 % Cc;
    const int h = hd / Dc, d = hd % Dc;
    const size_t dst = (((size_t)(b * Hc + h)) * Tc + t) * Dc + d;
    const unsigned short val = qkv[idx];
    if (which == 0) q[dst] = val;
    else if (which == 1) k[dst] = val;
    else v[dst] = val;
}

// ---------------- Flash attention: 4 waves/block, 1 wave per (b,h,q-tile) ----
__global__ __launch_bounds__(128)
void attn_flash_kernel(const unsigned short* __restrict__ qm,
                       const unsigned short* __restrict__ km,
                       const unsigned short* __restrict__ vm,
                       unsigned short* __restrict__ y) {
    __shared__ __align__(16) unsigned short p_lds[4][16 * 40];   // P 16x32 / wave
    __shared__ __align__(16) unsigned short vt_lds[4][64 * 40];  // V^T 64x32 / wave
    const int lane = threadIdx.x & 31;
    const int wid  = threadIdx.x >> 5;
    unsigned short* p_l  = p_lds[wid];
    unsigned short* vt_l = vt_lds[wid];

    const int nqt  = Tc / 16;
    const int gidx = blockIdx.x * 4 + wid;
    const int qt   = gidx % nqt;
    const int h    = (gidx / nqt) % Hc;
    const int b    = gidx / (nqt * Hc);
    const size_t hb = ((size_t)(b * Hc + h)) * Tc * Dc;
    const int lrow = lane & 15;
    const int kbl  = (lane >= 16) ? 8 : 0;
    const int hoff = (lane >= 16) ? 8 : 0;

    // Q fragments (16 x 64, split in two K=32 halves)
    const int qrow = qt * 16 + lrow;
    v16bf aq0 = ld_frag(&qm[hb + (size_t)qrow * Dc], kbl);        // d 0..31
    v16bf aq1 = ld_frag(&qm[hb + (size_t)qrow * Dc + 32], kbl);   // d 32..63

    float mrow[8], lsum[8];
    #pragma unroll
    for (int r = 0; r < 8; ++r) { mrow[r] = -__builtin_inff(); lsum[r] = 0.0f; }
    v8f o[4] = {};

    const int nk = (qt * 16 + 15) / 32 + 1;    // causal 32-key blocks
    for (int kb = 0; kb < nk; ++kb) {
        const int key0 = kb * 32;

        // --- scores: two 16x16 tiles over 32 keys ---
        v8f s[2];
        #pragma unroll
        for (int t = 0; t < 2; ++t) {
            const int key = key0 + t * 16 + lrow;
            v16bf bk0 = ld_frag(&km[hb + (size_t)key * Dc], kbl);
            v16bf bk1 = ld_frag(&km[hb + (size_t)key * Dc + 32], kbl);
            v8f z = {};
            z = __builtin_amdgcn_wmma_f32_16x16x32_bf16(false, aq0, false, bk0,
                                                        (short)0, z, false, false);
            z = __builtin_amdgcn_wmma_f32_16x16x32_bf16(false, aq1, false, bk1,
                                                        (short)0, z, false, false);
            s[t] = z;
        }

        // --- stage V^T (each lane owns one key row of 64 d values) ---
        {
            const unsigned short* vsrc = &vm[hb + (size_t)(key0 + lane) * Dc];
            union { uint4 q4[8]; unsigned short e[64]; } vr;
            #pragma unroll
            for (int j = 0; j < 8; ++j)
                vr.q4[j] = *reinterpret_cast<const uint4*>(&vsrc[j * 8]);
            #pragma unroll
            for (int j = 0; j < 64; ++j) vt_l[j * 40 + lane] = vr.e[j];
        }

        // --- online softmax over the 32 new keys ---
        float alpha[8];
        #pragma unroll
        for (int r = 0; r < 8; ++r) {
            const int rowg = qt * 16 + r + hoff;
            float s0 = s[0][r] * 0.125f;            // 1/sqrt(D)
            float s1 = s[1][r] * 0.125f;
            if (key0 + lrow > rowg)      s0 = -__builtin_inff();
            if (key0 + 16 + lrow > rowg) s1 = -__builtin_inff();
            float mx = redmax16(fmaxf(s0, s1));
            const float mnew = fmaxf(mrow[r], mx);
            const float a = expf(mrow[r] - mnew);
            const float p0 = expf(s0 - mnew);
            const float p1 = expf(s1 - mnew);
            lsum[r] = lsum[r] * a + redsum16(p0 + p1);
            mrow[r] = mnew;
            alpha[r] = a;
            p_l[(r + hoff) * 40 + lrow]      = f2bf(p0);
            p_l[(r + hoff) * 40 + 16 + lrow] = f2bf(p1);
        }
        lds_fence();   // per-wave LDS RAW sequencing (DS is in-order per wave)

        // --- rescale O and accumulate P @ V ---
        v16bf pf = ld_frag(&p_l[lrow * 40], kbl);
        #pragma unroll
        for (int dc = 0; dc < 4; ++dc) {
            #pragma unroll
            for (int r = 0; r < 8; ++r) o[dc][r] = o[dc][r] * alpha[r];
            v16bf vf = ld_frag(&vt_l[(dc * 16 + lrow) * 40], kbl);
            o[dc] = __builtin_amdgcn_wmma_f32_16x16x32_bf16(
                false, pf, false, vf, (short)0, o[dc], false, false);
        }
        lds_fence();
    }

    // --- normalize and write y[B,T,C] bf16 ---
    #pragma unroll
    for (int r = 0; r < 8; ++r) {
        const int rowg = qt * 16 + r + hoff;
        const float inv = 1.0f / lsum[r];
        #pragma unroll
        for (int dc = 0; dc < 4; ++dc) {
            y[((size_t)(b * Tc + rowg)) * Cc + h * Dc + dc * 16 + lrow] =
                f2bf(o[dc][r] * inv);
        }
    }
}

// ---------------- host launch ----------------
extern "C" void kernel_launch(void* const* d_in, const int* in_sizes, int n_in,
                              void* d_out, int out_size, void* d_ws, size_t ws_size,
                              hipStream_t stream) {
    const float* x      = (const float*)d_in[0];
    const float* ln1_w  = (const float*)d_in[1];
    const float* ln1_b  = (const float*)d_in[2];
    const float* Wqkv   = (const float*)d_in[3];
    const float* bqkv   = (const float*)d_in[4];
    const float* Wo     = (const float*)d_in[5];
    const float* bo     = (const float*)d_in[6];
    const float* ln2_w  = (const float*)d_in[7];
    const float* ln2_b  = (const float*)d_in[8];
    const float* Wfc    = (const float*)d_in[9];
    const float* bfc    = (const float*)d_in[10];
    const float* Wproj  = (const float*)d_in[11];
    const float* bproj  = (const float*)d_in[12];
    float* out = (float*)d_out;

    size_t off = 0;
    auto alloc = [&](size_t bytes) -> void* {
        off = (off + 255) & ~(size_t)255;
        void* p = (char*)d_ws + off;
        off += bytes;
        return p;
    };
    const size_t nWqkv = (size_t)Cc * 3 * Cc, nWo = (size_t)Cc * Cc;
    const size_t nWfc = (size_t)Cc * 4 * Cc, nWproj = (size_t)4 * Cc * Cc;
    unsigned short* wqkv_bf  = (unsigned short*)alloc(nWqkv * 2);
    unsigned short* wo_bf    = (unsigned short*)alloc(nWo * 2);
    unsigned short* wfc_bf   = (unsigned short*)alloc(nWfc * 2);
    unsigned short* wproj_bf = (unsigned short*)alloc(nWproj * 2);
    unsigned short* h1_bf    = (unsigned short*)alloc((size_t)Mrows * Cc * 2);
    unsigned short* qkv_bf   = (unsigned short*)alloc((size_t)Mrows * 3 * Cc * 2);
    unsigned short* q_bf     = (unsigned short*)alloc((size_t)Mrows * Cc * 2);
    unsigned short* k_bf     = (unsigned short*)alloc((size_t)Mrows * Cc * 2);
    unsigned short* v_bf     = (unsigned short*)alloc((size_t)Mrows * Cc * 2);
    unsigned short* y_bf     = (unsigned short*)alloc((size_t)Mrows * Cc * 2);
    float*          x1_f     = (float*)alloc((size_t)Mrows * Cc * 4);
    unsigned short* h2_bf    = (unsigned short*)alloc((size_t)Mrows * Cc * 2);
    unsigned short* hm_bf    = (unsigned short*)alloc((size_t)Mrows * 4 * Cc * 2);

    // 1) weights -> bf16
    to_bf16_kernel<<<(unsigned)(nWqkv / 256), 256, 0, stream>>>(Wqkv, wqkv_bf, nWqkv);
    to_bf16_kernel<<<(unsigned)(nWo / 256), 256, 0, stream>>>(Wo, wo_bf, nWo);
    to_bf16_kernel<<<(unsigned)(nWfc / 256), 256, 0, stream>>>(Wfc, wfc_bf, nWfc);
    to_bf16_kernel<<<(unsigned)(nWproj / 256), 256, 0, stream>>>(Wproj, wproj_bf, nWproj);

    // 2) LN1
    layernorm_bf16_kernel<<<Mrows, 256, 0, stream>>>(x, ln1_w, ln1_b, h1_bf);

    // 3) QKV GEMM -> bf16 [4096, 3072]
    gemm_bf16_wmma<1><<<dim3(3 * Cc / 128, Mrows / 128), 256, 0, stream>>>(
        h1_bf, wqkv_bf, bqkv, nullptr, qkv_bf, Mrows, 3 * Cc, Cc);

    // 4) split to [B,H,T,D]
    split_qkv_kernel<<<(unsigned)((size_t)Mrows * 3 * Cc / 256), 256, 0, stream>>>(
        qkv_bf, q_bf, k_bf, v_bf);

    // 5) flash attention -> y bf16 [B,T,C]
    attn_flash_kernel<<<Bc * Hc * (Tc / 16) / 4, 128, 0, stream>>>(
        q_bf, k_bf, v_bf, y_bf);

    // 6) Wo GEMM + residual x -> x1 f32
    gemm_bf16_wmma<3><<<dim3(Cc / 128, Mrows / 128), 256, 0, stream>>>(
        y_bf, wo_bf, bo, x, x1_f, Mrows, Cc, Cc);

    // 7) LN2
    layernorm_bf16_kernel<<<Mrows, 256, 0, stream>>>(x1_f, ln2_w, ln2_b, h2_bf);

    // 8) FC GEMM + exact GELU -> bf16 [4096, 4096]
    gemm_bf16_wmma<2><<<dim3(4 * Cc / 128, Mrows / 128), 256, 0, stream>>>(
        h2_bf, wfc_bf, bfc, nullptr, hm_bf, Mrows, 4 * Cc, Cc);

    // 9) Proj GEMM + residual x1 -> out f32
    gemm_bf16_wmma<3><<<dim3(Cc / 128, Mrows / 128), 256, 0, stream>>>(
        hm_bf, wproj_bf, bproj, x1_f, out, Mrows, Cc, 4 * Cc);

    (void)in_sizes; (void)n_in; (void)out_size; (void)ws_size;
}
